// Dual_LSTM_14688788152372
// MI455X (gfx1250) — matrix operand order repaired and checked by reading, not compiled, not run on hardware
//
#include <hip/hip_runtime.h>

typedef __attribute__((ext_vector_type(16))) __bf16 v16bf;
typedef __attribute__((ext_vector_type(8)))  __bf16 v8bf;
typedef __attribute__((ext_vector_type(8)))  float  v8f;

#define B_  512
#define T_  128
#define I_  128
#define H_  512
#define KC  640     // I_ + H_   (gate GEMM K)
#define G4  2048    // 4 * H_    (stacked gates f,i,o,cand)
#define KP  1024    // 2 * H_    (projection K)

// ---------------------------------------------------------------- helpers
__device__ __forceinline__ float fast_sigmoid(float x) {
    return 1.0f / (1.0f + __expf(-x));
}
__device__ __forceinline__ float fast_tanh(float x) {
    float e = __expf(-2.0f * fabsf(x));
    float t = (1.0f - e) / (1.0f + e);
    return x >= 0.0f ? t : -t;
}

// ---------------------------------------------------------------- prep kernels
__global__ void k_f32_to_bf16(const float* __restrict__ src,
                              __bf16* __restrict__ dst, int n) {
    int i = blockIdx.x * blockDim.x + threadIdx.x;
    int stride = gridDim.x * blockDim.x;
    for (; i < n; i += stride) dst[i] = (__bf16)src[i];
}

// W: [K, N] row-major fp32  ->  Wt: [N, K] row-major bf16
__global__ void k_transpose_bf16(const float* __restrict__ W,
                                 __bf16* __restrict__ Wt, int K, int N) {
    int i = blockIdx.x * blockDim.x + threadIdx.x;
    int stride = gridDim.x * blockDim.x;
    int total = K * N;
    for (; i < total; i += stride) {
        int n = i / K;
        int k = i - n * K;
        Wt[i] = (__bf16)W[(size_t)k * N + n];
    }
}

__global__ void k_zero_b32(unsigned int* __restrict__ p, int nwords) {
    int i = blockIdx.x * blockDim.x + threadIdx.x;
    int stride = gridDim.x * blockDim.x;
    for (; i < nwords; i += stride) p[i] = 0u;
}

// ---------------------------------------------------------------- LSTM cell step
// One launch = one timestep for BOTH cells (blockIdx.z selects s / l cell).
// Block: 256 threads = 8 waves; block tile = 32 rows x 64 hidden cols.
// Each wave: one 16x16 tile position, 4 gate accumulators (f,i,o,cand).
__global__ __launch_bounds__(256) void k_lstm_cell(
    const __bf16* __restrict__ xs,   // [B,T,I] bf16 (s-cell)
    const __bf16* __restrict__ xl,   // [B,T,I] bf16 (l-cell)
    const __bf16* __restrict__ sWt,  // [2048,640] bf16, N-major
    const __bf16* __restrict__ lWt,
    const float*  __restrict__ sb,   // [2048]
    const float*  __restrict__ lb,
    const __bf16* __restrict__ sh_in,  __bf16* __restrict__ sh_out,
    const __bf16* __restrict__ lh_in,  __bf16* __restrict__ lh_out,
    float* __restrict__ sc, float* __restrict__ lc,
    int t)
{
    const bool isL = (blockIdx.z != 0);
    const __bf16* x    = isL ? xl    : xs;
    const __bf16* Wt   = isL ? lWt   : sWt;
    const float*  bias = isL ? lb    : sb;
    const __bf16* h_in = isL ? lh_in : sh_in;
    __bf16* h_out      = isL ? lh_out: sh_out;
    float*  c          = isL ? lc    : sc;

    const int lane = threadIdx.x & 31;
    const int wave = threadIdx.x >> 5;
    const int half = lane >> 4;     // K-half select (this code's contiguous order, k = 16*half + s)
    const int l16  = lane & 15;
    const int m0 = blockIdx.x * 32 + (wave >> 2) * 16;   // batch-row tile
    const int n0 = blockIdx.y * 64 + (wave & 3) * 16;    // hidden-col tile

    v8f accF = {}, accI = {}, accO = {}, accC = {};

    // A fragment source: row = batch index; x part for K<128, h part after.
    const int arow = m0 + l16;
    const __bf16* xrow = x    + ((size_t)arow * T_ + t) * I_;
    const __bf16* hrow = h_in + (size_t)arow * H_;

    // B fragment source: lane l16 owns weight column (gate*512 + n0 + l16);
    // 16 consecutive K values, upper lane-half takes K+16.
    const int ncol = n0 + l16;
    const __bf16* WtF = Wt + (size_t)(0 * H_ + ncol) * KC + half * 16;
    const __bf16* WtI = Wt + (size_t)(1 * H_ + ncol) * KC + half * 16;
    const __bf16* WtO = Wt + (size_t)(2 * H_ + ncol) * KC + half * 16;
    const __bf16* WtG = Wt + (size_t)(3 * H_ + ncol) * KC + half * 16;

#pragma unroll
    for (int kt = 0; kt < KC / 32; ++kt) {
        const int k0 = kt * 32;
        const int kk = k0 + half * 16;
        v8bf alo, ahi;
        if (k0 < I_) {                       // resolved at compile time (unrolled)
            alo = *(const v8bf*)(xrow + kk);
            ahi = *(const v8bf*)(xrow + kk + 8);
        } else {
            alo = *(const v8bf*)(hrow + kk - I_);
            ahi = *(const v8bf*)(hrow + kk + 8 - I_);
        }
        v16bf afrag = __builtin_shufflevector(alo, ahi,
            0,1,2,3,4,5,6,7,8,9,10,11,12,13,14,15);

        v16bf bF = *(const v16bf*)(WtF + k0);
        v16bf bI = *(const v16bf*)(WtI + k0);
        v16bf bO = *(const v16bf*)(WtO + k0);
        v16bf bG = *(const v16bf*)(WtG + k0);

        accF = __builtin_amdgcn_wmma_f32_16x16x32_bf16(false, afrag, false, bF,
                                                       (short)0, accF, false, false);
        accI = __builtin_amdgcn_wmma_f32_16x16x32_bf16(false, afrag, false, bI,
                                                       (short)0, accI, false, false);
        accO = __builtin_amdgcn_wmma_f32_16x16x32_bf16(false, afrag, false, bO,
                                                       (short)0, accO, false, false);
        accC = __builtin_amdgcn_wmma_f32_16x16x32_bf16(false, afrag, false, bG,
                                                       (short)0, accC, false, false);
    }

    // Epilogue: gate math directly from accumulator layout
    // (VGPR r, lane-half h -> row m0 + 8*h + r, col n0 + l16).
    const float bF = bias[0 * H_ + ncol];
    const float bI = bias[1 * H_ + ncol];
    const float bO = bias[2 * H_ + ncol];
    const float bG = bias[3 * H_ + ncol];

#pragma unroll
    for (int r = 0; r < 8; ++r) {
        const int row = m0 + half * 8 + r;
        const size_t idx = (size_t)row * H_ + ncol;
        float fv = fast_sigmoid(accF[r] + bF);
        float iv = fast_sigmoid(accI[r] + bI);
        float ov = fast_sigmoid(accO[r] + bO);
        float gv = fast_tanh(accC[r] + bG);
        float cold = c[idx];
        float cnew = fv * cold + iv * gv;
        c[idx] = cnew;
        h_out[idx] = (__bf16)(ov * fast_tanh(cnew));
    }
}

// ---------------------------------------------------------------- output projection
// out = tanh([sh || lh] @ out_W + out_b)   [512,1024]x[1024,512]
__global__ __launch_bounds__(256) void k_output_proj(
    const __bf16* __restrict__ sh, const __bf16* __restrict__ lh,
    const __bf16* __restrict__ Wt,   // [512,1024] bf16, N-major
    const float*  __restrict__ ob,   // [512]
    float* __restrict__ out)         // [512,512]
{
    const int lane = threadIdx.x & 31;
    const int wave = threadIdx.x >> 5;
    const int half = lane >> 4;
    const int l16  = lane & 15;
    const int m0 = blockIdx.x * 32 + (wave >> 2) * 16;
    const int n0 = blockIdx.y * 64 + (wave & 3) * 16;

    v8f acc = {};
    const int arow = m0 + l16;
    const __bf16* srow = sh + (size_t)arow * H_;
    const __bf16* lrow = lh + (size_t)arow * H_;
    const int ncol = n0 + l16;
    const __bf16* wrow = Wt + (size_t)ncol * KP + half * 16;

#pragma unroll
    for (int kt = 0; kt < KP / 32; ++kt) {
        const int k0 = kt * 32;
        const int kk = k0 + half * 16;
        v8bf alo, ahi;
        if (k0 < H_) {
            alo = *(const v8bf*)(srow + kk);
            ahi = *(const v8bf*)(srow + kk + 8);
        } else {
            alo = *(const v8bf*)(lrow + kk - H_);
            ahi = *(const v8bf*)(lrow + kk + 8 - H_);
        }
        v16bf afrag = __builtin_shufflevector(alo, ahi,
            0,1,2,3,4,5,6,7,8,9,10,11,12,13,14,15);
        v16bf bfrag = *(const v16bf*)(wrow + k0);
        acc = __builtin_amdgcn_wmma_f32_16x16x32_bf16(false, afrag, false, bfrag,
                                                      (short)0, acc, false, false);
    }

    const float bb = ob[ncol];
#pragma unroll
    for (int r = 0; r < 8; ++r) {
        const int row = m0 + half * 8 + r;
        out[(size_t)row * H_ + ncol] = fast_tanh(acc[r] + bb);
    }
}

// ---------------------------------------------------------------- host wrapper
extern "C" void kernel_launch(void* const* d_in, const int* in_sizes, int n_in,
                              void* d_out, int out_size, void* d_ws, size_t ws_size,
                              hipStream_t stream) {
    (void)in_sizes; (void)n_in; (void)out_size; (void)ws_size;
    const float* s_inputs = (const float*)d_in[0];
    const float* l_inputs = (const float*)d_in[1];
    // d_in[2]=dow, d_in[3]=hour are unused by the reference math.
    const float* s_W   = (const float*)d_in[4];
    const float* s_b   = (const float*)d_in[5];
    const float* l_W   = (const float*)d_in[6];
    const float* l_b   = (const float*)d_in[7];
    const float* out_W = (const float*)d_in[8];
    const float* out_b = (const float*)d_in[9];
    float* out = (float*)d_out;

    char* ws = (char*)d_ws;
    // layout (bytes); zeroed region [sh0|lh0|sc|lc] is contiguous at front
    __bf16* sh0   = (__bf16*)(ws + 0);            //  512 KB
    __bf16* lh0   = (__bf16*)(ws + 524288);       //  512 KB
    float*  sc    = (float*) (ws + 1048576);      // 1 MB
    float*  lc    = (float*) (ws + 2097152);      // 1 MB
    __bf16* sh1   = (__bf16*)(ws + 3145728);      //  512 KB
    __bf16* lh1   = (__bf16*)(ws + 3670016);      //  512 KB
    __bf16* xs    = (__bf16*)(ws + 4194304);      // 16 MB
    __bf16* xl    = (__bf16*)(ws + 20971520);     // 16 MB
    __bf16* sWt   = (__bf16*)(ws + 37748736);     // 2.5 MB
    __bf16* lWt   = (__bf16*)(ws + 40370176);     // 2.5 MB
    __bf16* outWt = (__bf16*)(ws + 42991616);     // 1 MB
    // total = 44,040,192 bytes

    // 1) one-time precision/layout prep (bf16 casts + weight transposes)
    const int nX = B_ * T_ * I_;
    k_f32_to_bf16<<<1024, 256, 0, stream>>>(s_inputs, xs, nX);
    k_f32_to_bf16<<<1024, 256, 0, stream>>>(l_inputs, xl, nX);
    k_transpose_bf16<<<1024, 256, 0, stream>>>(s_W,   sWt,   KC, G4);
    k_transpose_bf16<<<1024, 256, 0, stream>>>(l_W,   lWt,   KC, G4);
    k_transpose_bf16<<<512,  256, 0, stream>>>(out_W, outWt, KP, H_);
    k_zero_b32<<<1024, 256, 0, stream>>>((unsigned int*)ws, 3145728 / 4);

    // 2) sequential scan: one launch per timestep, both cells via gridDim.z
    __bf16* shb[2] = {sh0, sh1};
    __bf16* lhb[2] = {lh0, lh1};
    dim3 grid(16, 8, 2);   // 512/32 row tiles, 512/64 col tiles, 2 cells
    for (int t = 0; t < T_; ++t) {
        k_lstm_cell<<<grid, 256, 0, stream>>>(
            xs, xl, sWt, lWt, s_b, l_b,
            shb[t & 1], shb[(t + 1) & 1],
            lhb[t & 1], lhb[(t + 1) & 1],
            sc, lc, t);
    }
    // after t=127 the final h lives in buffer 0

    // 3) output projection + tanh
    k_output_proj<<<dim3(16, 8, 1), 256, 0, stream>>>(sh0, lh0, outWt, out_b, out);
}
